// MultiHeadAttention_10101763080466
// MI455X (gfx1250) — matrix operand order repaired and checked by reading, not compiled, not run on hardware
//
#include <hip/hip_runtime.h>

typedef __attribute__((ext_vector_type(16))) __bf16 v16bf;
typedef __attribute__((ext_vector_type(8)))  float  v8f;
typedef __attribute__((ext_vector_type(4)))  float  v4f;
typedef __attribute__((ext_vector_type(4)))  __bf16 v4bf;
typedef __attribute__((ext_vector_type(8)))  __bf16 v8bf;

union Frag {
  unsigned int u[8];
  v16bf v;
};

static __device__ inline v8f wmma_bf16(const Frag& a, const Frag& b, v8f c) {
  // D = A(16x32 bf16) x B(32x16 bf16) + C(16x16 f32)
  return __builtin_amdgcn_wmma_f32_16x16x32_bf16(
      false, a.v, false, b.v, (short)0, c, false, false);
}

// vectorized load of 4 elems + convert to bf16
static __device__ inline v4bf cvt4(const float* src) {
  v4f t = *(const v4f*)src;
  v4bf r;
#pragma unroll
  for (int j = 0; j < 4; ++j) r[j] = (__bf16)t[j];
  return r;
}
static __device__ inline v4bf cvt4(const __bf16* src) {
  return *(const v4bf*)src;
}

// ---------------------------------------------------------------------------
// Tiled GEMM with bias:  C[M,N] = A[M,K] * W[K,N] + bias[N]
// Block tile 128x128, BK=32, 8 waves (256 thr), wave tile 64x32 (4x2 WMMA).
// Double-buffered LDS overlaps global staging of tile k+1 with the 8 WMMAs
// of tile k.  Global addresses: uniform 64-bit base + invariant 32-bit
// per-lane offset -> saddr-form loads, no per-iter 64-bit VALU math.
// ---------------------------------------------------------------------------
#define BM 128
#define BN 128
#define BK 32

template <typename TIN, typename TOUT>
__global__ __launch_bounds__(256)
void gemm_bias_wmma(const TIN* __restrict__ A, const float* __restrict__ W,
                    const float* __restrict__ bias, TOUT* __restrict__ C,
                    int M, int N, int K) {
  __shared__ __bf16 lA[2][BM * BK];  // [m][k]
  __shared__ __bf16 lB[2][BN * BK];  // [n][k] (W transposed on store)

  const int tid  = threadIdx.x;
  const int wave = tid >> 5;
  const int lane = tid & 31;
  const int half = lane >> 4;
  const int l16  = lane & 15;
  const int wm   = (wave >> 2) * 64;
  const int wn   = (wave & 3) * 32;
  const int m0   = blockIdx.y * BM;
  const int n0   = blockIdx.x * BN;

  // uniform bases (scalar); per-lane 32-bit offsets (loop-invariant)
  const TIN*   Arow = A + (size_t)m0 * K;
  const float* Wrow = W + n0;
  int aoff[4], woff[4];
#pragma unroll
  for (int i = 0; i < 4; ++i) {
    int q = i * 256 + tid;
    aoff[i] = (q >> 3) * K + (q & 7) * 4;   // m*K + kq
    woff[i] = (q >> 5) * N + (q & 31) * 4;  // k*N + nq
  }

  auto stage = [&](int k0, int buf) {
    const TIN*   Ab = Arow + k0;
    const float* Wb = Wrow + (size_t)k0 * N;
#pragma unroll
    for (int i = 0; i < 4; ++i) {
      int q = i * 256 + tid;
      int m = q >> 3, kq = (q & 7) * 4;
      v4bf t = cvt4(&Ab[aoff[i]]);
      *(v4bf*)&lA[buf][m * BK + kq] = t;
    }
#pragma unroll
    for (int i = 0; i < 4; ++i) {
      int q = i * 256 + tid;
      int k = q >> 5, nq = (q & 31) * 4;
      v4f w = *(const v4f*)&Wb[woff[i]];
#pragma unroll
      for (int j = 0; j < 4; ++j) lB[buf][(nq + j) * BK + k] = (__bf16)w[j];
    }
  };

  v8f acc[4][2] = {};

  stage(0, 0);
  const int nk = K / BK;
  for (int kt = 0; kt < nk; ++kt) {
    const int cur = kt & 1;
    __syncthreads();
    if (kt + 1 < nk) {
      __builtin_prefetch(&Arow[(size_t)(tid >> 1) * K + (kt + 1) * BK], 0, 1);
      stage((kt + 1) * BK, cur ^ 1);
    }

    Frag aF[4];
#pragma unroll
    for (int mt = 0; mt < 4; ++mt) {
      const unsigned int* p =
          (const unsigned int*)&lA[cur][(wm + mt * 16 + l16) * BK];
#pragma unroll
      for (int j = 0; j < 8; ++j)
        aF[mt].u[j] = p[((j < 4) ? j : j + 4) + 4 * half];
    }
#pragma unroll
    for (int nt = 0; nt < 2; ++nt) {
      Frag bF;
      const unsigned int* p =
          (const unsigned int*)&lB[cur][(wn + nt * 16 + l16) * BK];
#pragma unroll
      for (int j = 0; j < 8; ++j) bF.u[j] = p[((j < 4) ? j : j + 4) + 4 * half];
#pragma unroll
      for (int mt = 0; mt < 4; ++mt)
        acc[mt][nt] = wmma_bf16(aF[mt], bF, acc[mt][nt]);
    }
  }

  TOUT* Cb = C + (size_t)m0 * N + n0;
#pragma unroll
  for (int mt = 0; mt < 4; ++mt)
#pragma unroll
    for (int nt = 0; nt < 2; ++nt) {
      int cn = wn + nt * 16 + l16;
      float bv = bias[n0 + cn];
#pragma unroll
      for (int r = 0; r < 8; ++r) {
        int cm = wm + mt * 16 + r + 8 * half;
        Cb[cm * N + cn] = (TOUT)(acc[mt][nt][r] + bv);
      }
    }
}

// ---------------------------------------------------------------------------
// Flash attention, S^T formulation: one block = (b, h, 64 q-rows), 4 waves.
// S^T = V_tile(64x64) . Q^T  -> both operands row-major in LDS; each
// q-column lives in a lane pair {L, L^16}: softmax = per-lane register
// reduction + a single shfl_xor(16).  Softmax in log2 domain (one fused
// scale*log2e constant, v_exp_f32 per element).  lP/lVt rows padded to 72.
// ---------------------------------------------------------------------------
#define DQT 64
#define DKT 64
#define DD  64
#define DKTP 72

__global__ __launch_bounds__(128)
void attn_wmma(const __bf16* __restrict__ Q, const __bf16* __restrict__ V,
               __bf16* __restrict__ Ctx, int SQ, int SKV, int F) {
  __shared__ __bf16 lQ[DQT * DD];        // [q][d]
  __shared__ __bf16 lV[DKT * DD];        // [kv][d]
  __shared__ __bf16 lVt[DD * DKTP];      // [d][kv]  (padded)
  __shared__ __bf16 lP[4][16 * DKTP];    // per-wave P tile [q16][kv64]
  __shared__ float  lCorr[4][16];        // per-wave per-q correction
  __shared__ float  lInv[4][16];         // per-wave per-q 1/l

  const int tid  = threadIdx.x;
  const int wave = tid >> 5;
  const int lane = tid & 31;
  const int half = lane >> 4;
  const int l16  = lane & 15;
  const int qb   = blockIdx.x * DQT;
  const int h    = blockIdx.y;
  const int b    = blockIdx.z;
  const int c0 = h * DD;
  // softmax in log2 domain: fold 1/sqrt(64) and log2(e) into one constant
  const float SL2E = 0.125f * 1.4426950408889634f;

  // uniform bases + loop-invariant 32-bit per-lane offsets
  const __bf16* Qbase = Q + ((size_t)b * SQ + qb) * F + c0;
  const __bf16* Vrow  = V + (size_t)b * SKV * F + c0;
  int goff[4], lvo[4], lvto[4];
#pragma unroll
  for (int i = 0; i < 4; ++i) {
    int o = i * 128 + tid;
    int k = o >> 3, d8 = (o & 7) * 8;
    goff[i] = k * F + d8;          // global row*F + col (32-bit)
    lvo[i]  = k * DD + d8;         // lV  [kv][d]
    lvto[i] = d8 * DKTP + k;       // lVt [d][kv], +j*DKTP immediates
  }

  // stage Q tile once
#pragma unroll
  for (int i = 0; i < 4; ++i) {
    int o = i * 128 + tid;
    int q = o >> 3, d8 = (o & 7) * 8;
    *(v8bf*)&lQ[q * DD + d8] = *(const v8bf*)&Qbase[goff[i]];
  }
  __syncthreads();

  // Q^T B-fragments are kv-invariant: hoist (lane n = q, K pairs over d)
  Frag qB[2];
#pragma unroll
  for (int ks = 0; ks < 2; ++ks) {
    const unsigned int* p =
        (const unsigned int*)&lQ[(wave * 16 + l16) * DD + ks * 32];
#pragma unroll
    for (int j = 0; j < 8; ++j) qB[ks].u[j] = p[((j < 4) ? j : j + 4) + 4 * half];
  }

  float mrun = -1e30f, lrun = 0.f;
  v8f accC[4] = {};   // ctx: rows q (r+8*half), cols d (nt*16+l16)

  for (int kv0 = 0; kv0 < SKV; kv0 += DKT) {
    __syncthreads();  // previous iteration's lV/lVt reads complete
    const __bf16* vb = Vrow + (size_t)kv0 * F;  // uniform advance
#pragma unroll
    for (int i = 0; i < 4; ++i) {
      v8bf t = *(const v8bf*)&vb[goff[i]];
      *(v8bf*)&lV[lvo[i]] = t;
#pragma unroll
      for (int j = 0; j < 8; ++j) lVt[lvto[i] + j * DKTP] = t[j];
    }
    __syncthreads();

    // ---- S^T = V(64x64) . Q^T(64x16): A = V kv-rows, B = Q^T ----
    v8f accS[4] = {};   // [mt]: rows kv (mt*16 + r + 8*half), col q = l16
#pragma unroll
    for (int ks = 0; ks < 2; ++ks) {
#pragma unroll
      for (int mt = 0; mt < 4; ++mt) {
        Frag aF;
        const unsigned int* pa =
            (const unsigned int*)&lV[(mt * 16 + l16) * DD + ks * 32];
#pragma unroll
        for (int j = 0; j < 8; ++j)
          aF.u[j] = pa[((j < 4) ? j : j + 4) + 4 * half];
        accS[mt] = wmma_bf16(aF, qB[ks], accS[mt]);
      }
    }

    // ---- online softmax (log2 domain), 2 shuffles per iter ----
    float mx = -1e30f;
#pragma unroll
    for (int mt = 0; mt < 4; ++mt)
#pragma unroll
      for (int r = 0; r < 8; ++r) mx = fmaxf(mx, accS[mt][r]);
    mx *= SL2E;
    mx = fmaxf(mx, __shfl_xor(mx, 16, 32));
    float mnew = fmaxf(mrun, mx);
    float corr = __builtin_amdgcn_exp2f(mrun - mnew);
    mrun = mnew;

    float rs = 0.f;
#pragma unroll
    for (int mt = 0; mt < 4; ++mt) {
      v8bf pv;
#pragma unroll
      for (int r = 0; r < 8; ++r) {
        float p = __builtin_amdgcn_exp2f(accS[mt][r] * SL2E - mnew);
        rs += p;
        pv[r] = (__bf16)p;
      }
      // P^T D-layout -> row-major [q][kv]: 8 consecutive kv at fixed q
      *(v8bf*)&lP[wave][l16 * DKTP + mt * 16 + 8 * half] = pv;
    }
    rs += __shfl_xor(rs, 16, 32);
    lrun = lrun * corr + rs;
    if (!half) lCorr[wave][l16] = corr;

    // wave-local RAW (LDS in-order per wave): wait stores, block reordering
    asm volatile("s_wait_dscnt 0" ::: "memory");
    __builtin_amdgcn_wave_barrier();

    // redistribute per-q corr to ctx rows (q = r + 8*half)
    v4f cA = *(v4f*)&lCorr[wave][8 * half];
    v4f cB = *(v4f*)&lCorr[wave][8 * half + 4];
#pragma unroll
    for (int nt = 0; nt < 4; ++nt)
#pragma unroll
      for (int r = 0; r < 8; ++r)
        accC[nt][r] *= (r < 4) ? cA[r] : cB[r - 4];

    // ---- Ctx += P (16x64) . V (64x64) ----
#pragma unroll
    for (int ks = 0; ks < 2; ++ks) {
      Frag aF;
      const unsigned int* pa =
          (const unsigned int*)&lP[wave][l16 * DKTP + ks * 32];
#pragma unroll
      for (int j = 0; j < 8; ++j)
        aF.u[j] = pa[((j < 4) ? j : j + 4) + 4 * half];
#pragma unroll
      for (int nt = 0; nt < 4; ++nt) {
        Frag bF;
        const unsigned int* pb =
            (const unsigned int*)&lVt[(nt * 16 + l16) * DKTP + ks * 32];
#pragma unroll
        for (int j = 0; j < 8; ++j) bF.u[j] = pb[((j < 4) ? j : j + 4) + 4 * half];
        accC[nt] = wmma_bf16(aF, bF, accC[nt]);
      }
    }
  }

  // normalize: per-q 1/l redistributed to ctx rows, then store
  if (!half) lInv[wave][l16] = 1.0f / lrun;
  asm volatile("s_wait_dscnt 0" ::: "memory");
  __builtin_amdgcn_wave_barrier();
  v4f iA = *(v4f*)&lInv[wave][8 * half];
  v4f iB = *(v4f*)&lInv[wave][8 * half + 4];
  __bf16* Cbase = Ctx + ((size_t)b * SQ + qb) * F + c0;
#pragma unroll
  for (int nt = 0; nt < 4; ++nt)
#pragma unroll
    for (int r = 0; r < 8; ++r) {
      int row = wave * 16 + r + 8 * half;
      float inv = (r < 4) ? iA[r] : iB[r - 4];
      Cbase[row * F + nt * 16 + l16] = (__bf16)(accC[nt][r] * inv);
    }
}

// ---------------------------------------------------------------------------
extern "C" void kernel_launch(void* const* d_in, const int* in_sizes, int n_in,
                              void* d_out, int out_size, void* d_ws,
                              size_t ws_size, hipStream_t stream) {
  (void)in_sizes; (void)n_in; (void)out_size; (void)ws_size;
  const int B = 4, SQ = 1024, SKV = 1024, DIN = 1024, F = 1024, H = 16;
  const int M = B * SQ;  // 4096

  const float* features = (const float*)d_in[0];  // [B,SKV,DIN]
  const float* state    = (const float*)d_in[1];  // [B,SQ,DIN]
  const float* Wq = (const float*)d_in[2];
  const float* bq = (const float*)d_in[3];
  const float* Wv = (const float*)d_in[4];
  const float* bv = (const float*)d_in[5];
  const float* Wo = (const float*)d_in[6];
  const float* bo = (const float*)d_in[7];
  float* out = (float*)d_out;

  __bf16* Qbuf = (__bf16*)d_ws;                  // 8 MB
  __bf16* Vbuf = Qbuf + (size_t)M * F;           // 8 MB
  __bf16* Cbuf = Vbuf + (size_t)M * F;           // 8 MB

  dim3 gg(F / BN, M / BM);  // (8, 32)

  gemm_bias_wmma<float, __bf16><<<gg, 256, 0, stream>>>(
      state, Wq, bq, Qbuf, M, F, DIN);
  gemm_bias_wmma<float, __bf16><<<gg, 256, 0, stream>>>(
      features, Wv, bv, Vbuf, M, F, DIN);

  attn_wmma<<<dim3(SQ / DQT, H, B), 128, 0, stream>>>(
      Qbuf, Vbuf, Cbuf, SQ, SKV, F);

  gemm_bias_wmma<__bf16, float><<<gg, 256, 0, stream>>>(
      Cbuf, Wo, bo, out, M, F, F);
}
